// NaViT_10376640987904
// MI455X (gfx1250) — matrix operand order repaired and checked by reading, not compiled, not run on hardware
//
#include <hip/hip_runtime.h>
#include <math.h>

typedef __bf16 bf16_t;
typedef __attribute__((ext_vector_type(16))) __bf16 v16bf;
typedef __attribute__((ext_vector_type(8)))  __bf16 v8bf;
typedef __attribute__((ext_vector_type(4)))  __bf16 v4bf;
typedef __attribute__((ext_vector_type(8)))  float   v8f;
typedef __attribute__((ext_vector_type(4)))  float   v4f;

#define E_DIM 768
#define NHEAD 12
#define HDIM  64
#define NTOK  196
#define NPAD  224
#define BATCH 32
#define MROWS (BATCH * NTOK)   /* 6272 */
#define FFDIM 3072

__device__ __forceinline__ bf16_t f2bf(float f) {
  unsigned u = __builtin_bit_cast(unsigned, f);
  unsigned r = u + 0x7FFFu + ((u >> 16) & 1u);
  unsigned short h = (unsigned short)(r >> 16);
  return __builtin_bit_cast(bf16_t, h);
}
__device__ __forceinline__ float bf2f(bf16_t h) {
  unsigned short s = __builtin_bit_cast(unsigned short, h);
  unsigned u = ((unsigned)s) << 16;
  return __builtin_bit_cast(float, u);
}
__device__ __forceinline__ v8f v8f_zero() {
  v8f z = {0.f,0.f,0.f,0.f,0.f,0.f,0.f,0.f}; return z;
}
__device__ __forceinline__ v8bf v8bf_zero() {
  bf16_t z = f2bf(0.0f);
  v8bf r = {z,z,z,z,z,z,z,z}; return r;
}

union Frag16 { v16bf v; v8bf h[2]; };

// ---------------------------------------------------------------------------
// fp32 -> bf16 bulk conversion (vector-4), run once per launch on all weights
// ---------------------------------------------------------------------------
__global__ __launch_bounds__(256)
void cvt_f32_bf16(const float* __restrict__ src, bf16_t* __restrict__ dst, long n4)
{
  long i = (long)blockIdx.x * 256 + threadIdx.x;
  if (i >= n4) return;
  v4f f = ((const v4f*)src)[i];
  v4bf o;
#pragma unroll
  for (int j = 0; j < 4; ++j) o[j] = f2bf(f[j]);
  ((v4bf*)dst)[i] = o;
}

// ---------------------------------------------------------------------------
// GEMM: C[M,N] = A[M,K](bf16) @ Wb[N,K](bf16)^T  (+bias, GELU, residual)
// 256 thr = 8 waves (2 M x 4 N); block tile 128x256; wave tile 64x64.
// Per K-step(32): 16x global_load_b128 feeding 16x v_wmma (bf16, f32 acc).
// BOUNDED=0: assumes M%128==0 && N%256==0 (no guards in hot loop).
// ---------------------------------------------------------------------------
template <int BOUNDED>
__global__ __launch_bounds__(256)
void gemm_wmma_bf16(const bf16_t* __restrict__ A, const bf16_t* __restrict__ Wb,
                    const float* __restrict__ bias, const float* __restrict__ resid,
                    float* __restrict__ Cf, bf16_t* __restrict__ Cb,
                    int M, int N, int K, int gelu)
{
  const int tid  = threadIdx.x;
  const int lane = tid & 31;
  const int wave = tid >> 5;
  const int l16  = lane & 15;
  const int half = lane >> 4;
  const int m0 = blockIdx.y * 128 + (wave & 1) * 64;
  const int n0 = blockIdx.x * 256 + (wave >> 1) * 64;

  v8f acc[4][4];
#pragma unroll
  for (int i = 0; i < 4; ++i)
#pragma unroll
    for (int j = 0; j < 4; ++j) acc[i][j] = v8f_zero();

  // Per-lane base pointers (A layout: half -> K {0..7,16..23}/{8..15,24..31};
  // B layout: lane col = l16, half -> K {0..7,16..23}/{8..15,24..31}).
  const bf16_t* ap[4];
  const bf16_t* bp[4];
#pragma unroll
  for (int mt = 0; mt < 4; ++mt) {
    int row = m0 + mt * 16 + l16;
    int rc  = BOUNDED ? (row < M ? row : 0) : row;
    ap[mt] = A + (size_t)rc * K + half * 8;
  }
#pragma unroll
  for (int nt = 0; nt < 4; ++nt) {
    int col = n0 + nt * 16 + l16;
    int cc  = BOUNDED ? (col < N ? col : 0) : col;
    bp[nt] = Wb + (size_t)cc * K + half * 8;
  }

  for (int kb = 0; kb < K; kb += 32) {
    Frag16 af[4], bfr[4];
#pragma unroll
    for (int mt = 0; mt < 4; ++mt) {
      af[mt].h[0] = *(const v8bf*)(ap[mt]);
      af[mt].h[1] = *(const v8bf*)(ap[mt] + 16);
      ap[mt] += 32;
    }
#pragma unroll
    for (int nt = 0; nt < 4; ++nt) {
      bfr[nt].h[0] = *(const v8bf*)(bp[nt]);
      bfr[nt].h[1] = *(const v8bf*)(bp[nt] + 16);
      bp[nt] += 32;
    }
    if (BOUNDED) {  // zero out-of-range fragments (clamped rows/cols loaded junk)
#pragma unroll
      for (int mt = 0; mt < 4; ++mt)
        if (m0 + mt * 16 + l16 >= M) { af[mt].h[0] = v8bf_zero(); af[mt].h[1] = v8bf_zero(); }
#pragma unroll
      for (int nt = 0; nt < 4; ++nt)
        if (n0 + nt * 16 + l16 >= N) { bfr[nt].h[0] = v8bf_zero(); bfr[nt].h[1] = v8bf_zero(); }
    }
#pragma unroll
    for (int mt = 0; mt < 4; ++mt)
#pragma unroll
      for (int nt = 0; nt < 4; ++nt)
        acc[mt][nt] = __builtin_amdgcn_wmma_f32_16x16x32_bf16(
            false, af[mt].v, false, bfr[nt].v, (short)0, acc[mt][nt], false, false);
  }

  // epilogue: C layout (VGPR r, lane): M = r + 8*half, N = l16
#pragma unroll
  for (int nt = 0; nt < 4; ++nt) {
    int col = n0 + nt * 16 + l16;
    if (BOUNDED && col >= N) continue;
    float bv = bias ? bias[col] : 0.0f;
#pragma unroll
    for (int mt = 0; mt < 4; ++mt) {
#pragma unroll
      for (int r = 0; r < 8; ++r) {
        int row = m0 + mt * 16 + r + half * 8;
        if (BOUNDED && row >= M) continue;
        float v = acc[mt][nt][r] + bv;
        if (gelu) v = 0.5f * v * (1.0f + erff(v * 0.70710678118654752f));
        size_t idx = (size_t)row * N + col;
        if (resid) v += resid[idx];
        if (Cf) Cf[idx] = v;
        if (Cb) Cb[idx] = f2bf(v);
      }
    }
  }
}

// ---------------------------------------------------------------------------
// Patch gather: x[32,3,224,224] -> bf16 patches [6272, 768] (c*256+py*16+px)
// ---------------------------------------------------------------------------
__global__ __launch_bounds__(256)
void patch_gather(const float* __restrict__ x, bf16_t* __restrict__ out)
{
  int m = blockIdx.x;
  int b = m / NTOK, n = m % NTOK;
  int pr = n / 14, pc = n % 14;
#pragma unroll
  for (int i = 0; i < 3; ++i) {
    int k = threadIdx.x + i * 256;
    int ch = k >> 8, py = (k >> 4) & 15, px = k & 15;
    float v = x[(((size_t)b * 3 + ch) * 224 + pr * 16 + py) * 224 + pc * 16 + px];
    out[(size_t)m * E_DIM + k] = f2bf(v);
  }
}

// ---------------------------------------------------------------------------
// Factorized positional embedding add (in-place on f32 tok)
// ---------------------------------------------------------------------------
__global__ __launch_bounds__(256)
void pos_add(float* __restrict__ tok, const float* __restrict__ he,
             const float* __restrict__ we)
{
  int m = blockIdx.x; int n = m % NTOK;
  int pr = n / 14, pc = n % 14;
#pragma unroll
  for (int i = 0; i < 3; ++i) {
    int e = threadIdx.x + i * 256;
    float add = (e < 384) ? he[pr * 384 + e] : we[pc * 384 + (e - 384)];
    tok[(size_t)m * E_DIM + e] += add;
  }
}

// ---------------------------------------------------------------------------
// LayerNorm over E=768 (one block per row), output bf16
// ---------------------------------------------------------------------------
__global__ __launch_bounds__(256)
void layernorm768(const float* __restrict__ X, const float* __restrict__ w,
                  const float* __restrict__ bvec, bf16_t* __restrict__ Y)
{
  int row = blockIdx.x;
  const float* xr = X + (size_t)row * E_DIM;
  float v0 = xr[threadIdx.x], v1 = xr[threadIdx.x + 256], v2 = xr[threadIdx.x + 512];
  float s1 = v0 + v1 + v2;
  float s2 = v0 * v0 + v1 * v1 + v2 * v2;
#pragma unroll
  for (int m = 16; m >= 1; m >>= 1) { s1 += __shfl_xor(s1, m, 32); s2 += __shfl_xor(s2, m, 32); }
  __shared__ float r1[8], r2[8];
  int wv = threadIdx.x >> 5, ln = threadIdx.x & 31;
  if (ln == 0) { r1[wv] = s1; r2[wv] = s2; }
  __syncthreads();
  float t1 = 0.f, t2 = 0.f;
#pragma unroll
  for (int i = 0; i < 8; ++i) { t1 += r1[i]; t2 += r2[i]; }
  float mean = t1 * (1.0f / 768.0f);
  float var  = t2 * (1.0f / 768.0f) - mean * mean;
  float rs = rsqrtf(var + 1e-5f);
  bf16_t* yr = Y + (size_t)row * E_DIM;
  yr[threadIdx.x]       = f2bf((v0 - mean) * rs * w[threadIdx.x]       + bvec[threadIdx.x]);
  yr[threadIdx.x + 256] = f2bf((v1 - mean) * rs * w[threadIdx.x + 256] + bvec[threadIdx.x + 256]);
  yr[threadIdx.x + 512] = f2bf((v2 - mean) * rs * w[threadIdx.x + 512] + bvec[threadIdx.x + 512]);
}

// ---------------------------------------------------------------------------
// Per-head q/k LayerNorm + layout: q,k -> [BH, 224, 64] bf16 (zero-padded),
// v -> transposed [BH, 64, 224] bf16 (zero-padded). One wave handles a head.
// ---------------------------------------------------------------------------
__device__ __forceinline__ float wave_sum32(float v) {
#pragma unroll
  for (int m = 16; m >= 1; m >>= 1) v += __shfl_xor(v, m, 32);
  return v;
}

__global__ __launch_bounds__(256)
void qkv_headnorm(const float* __restrict__ qkv,
                  const float* __restrict__ qw, const float* __restrict__ qb,
                  const float* __restrict__ kw, const float* __restrict__ kbv,
                  bf16_t* __restrict__ qn, bf16_t* __restrict__ kn,
                  bf16_t* __restrict__ vnT)
{
  int bj = blockIdx.x;
  int b = bj / NPAD, j = bj % NPAD;
  int wv = threadIdx.x >> 5, lane = threadIdx.x & 31;
  int d0 = 2 * lane, d1 = 2 * lane + 1;
  for (int h = wv; h < NHEAD; h += 8) {
    size_t hb = (size_t)(b * NHEAD + h);
    size_t oq = (hb * NPAD + j) * HDIM;
    if (j >= NTOK) {
      bf16_t z = f2bf(0.0f);
      qn[oq + d0] = z; qn[oq + d1] = z;
      kn[oq + d0] = z; kn[oq + d1] = z;
      vnT[(hb * HDIM + d0) * NPAD + j] = z;
      vnT[(hb * HDIM + d1) * NPAD + j] = z;
    } else {
      size_t m = (size_t)b * NTOK + j;
      const float* qp = qkv + m * (3 * E_DIM) + h * HDIM;
      const float* kp = qp + E_DIM;
      const float* vp = qp + 2 * E_DIM;
      float q0 = qp[d0], q1 = qp[d1];
      float k0 = kp[d0], k1 = kp[d1];
      float mq = wave_sum32(q0 + q1) * (1.0f / 64.0f);
      float dq0 = q0 - mq, dq1 = q1 - mq;
      float vq = wave_sum32(dq0 * dq0 + dq1 * dq1) * (1.0f / 64.0f);
      float rq = rsqrtf(vq + 1e-5f);
      qn[oq + d0] = f2bf(dq0 * rq * qw[d0] + qb[d0]);
      qn[oq + d1] = f2bf(dq1 * rq * qw[d1] + qb[d1]);
      float mk = wave_sum32(k0 + k1) * (1.0f / 64.0f);
      float dk0 = k0 - mk, dk1 = k1 - mk;
      float vk = wave_sum32(dk0 * dk0 + dk1 * dk1) * (1.0f / 64.0f);
      float rk = rsqrtf(vk + 1e-5f);
      kn[oq + d0] = f2bf(dk0 * rk * kw[d0] + kbv[d0]);
      kn[oq + d1] = f2bf(dk1 * rk * kw[d1] + kbv[d1]);
      vnT[(hb * HDIM + d0) * NPAD + j] = f2bf(vp[d0]);
      vnT[(hb * HDIM + d1) * NPAD + j] = f2bf(vp[d1]);
    }
  }
}

// ---------------------------------------------------------------------------
// Fused attention per (b,h): S = q k^T (WMMA, K=64), softmax in-register,
// O = P V (WMMA, K=224 padded) via per-wave LDS P staging. Block = 4 waves.
// ---------------------------------------------------------------------------
__global__ __launch_bounds__(128)
void attention_wmma(const bf16_t* __restrict__ qn, const bf16_t* __restrict__ kn,
                    const bf16_t* __restrict__ vnT, bf16_t* __restrict__ ob)
{
  __shared__ bf16_t ks[NPAD * HDIM];   // K: [token][d]
  __shared__ bf16_t vs[HDIM * NPAD];   // V^T: [d][token]
  __shared__ bf16_t pt[4][16 * 32];    // per-wave P staging tile
  int h = blockIdx.x, b = blockIdx.y;
  size_t hb = (size_t)(b * NHEAD + h);
  int tid = threadIdx.x;
  {
    const v8bf* gk = (const v8bf*)(kn + hb * (size_t)NPAD * HDIM);
    const v8bf* gv = (const v8bf*)(vnT + hb * (size_t)HDIM * NPAD);
    v8bf* sk = (v8bf*)ks;
    v8bf* sv = (v8bf*)vs;
    for (int i = tid; i < NPAD * HDIM / 8; i += 128) { sk[i] = gk[i]; sv[i] = gv[i]; }
  }
  __syncthreads();
  int lane = tid & 31, wv = tid >> 5;
  int l16 = lane & 15, half = lane >> 4;
  const float scale = 0.125f;  // 64^-0.5

  for (int mt = wv; mt < 13; mt += 4) {
    Frag16 qf[2];
    const bf16_t* qrow = qn + (hb * NPAD + mt * 16 + l16) * HDIM;
#pragma unroll
    for (int kq = 0; kq < 2; ++kq) {
      const bf16_t* p = qrow + kq * 32 + half * 8;
      qf[kq].h[0] = *(const v8bf*)p;
      qf[kq].h[1] = *(const v8bf*)(p + 16);
    }
    v8f s[14];
#pragma unroll
    for (int jt = 0; jt < 14; ++jt) s[jt] = v8f_zero();
#pragma unroll
    for (int jt = 0; jt < 14; ++jt) {
#pragma unroll
      for (int kq = 0; kq < 2; ++kq) {
        Frag16 bfr;
        const bf16_t* kp = ks + (jt * 16 + l16) * HDIM + kq * 32 + half * 8;
        bfr.h[0] = *(const v8bf*)kp;
        bfr.h[1] = *(const v8bf*)(kp + 16);
        s[jt] = __builtin_amdgcn_wmma_f32_16x16x32_bf16(
            false, qf[kq].v, false, bfr.v, (short)0, s[jt], false, false);
      }
    }
    float mx[8], inv[8];
#pragma unroll
    for (int r = 0; r < 8; ++r) {
      float lm = -1e30f;
#pragma unroll
      for (int jt = 0; jt < 14; ++jt) {
        int col = jt * 16 + l16;
        float v = s[jt][r] * scale;
        if (col < NTOK) lm = fmaxf(lm, v);
      }
#pragma unroll
      for (int msk = 8; msk >= 1; msk >>= 1) lm = fmaxf(lm, __shfl_xor(lm, msk, 32));
      mx[r] = lm;
    }
#pragma unroll
    for (int r = 0; r < 8; ++r) {
      float ls = 0.f;
#pragma unroll
      for (int jt = 0; jt < 14; ++jt) {
        int col = jt * 16 + l16;
        float e = (col < NTOK) ? __expf(s[jt][r] * scale - mx[r]) : 0.f;
        s[jt][r] = e;
        ls += e;
      }
#pragma unroll
      for (int msk = 8; msk >= 1; msk >>= 1) ls += __shfl_xor(ls, msk, 32);
      inv[r] = 1.0f / ls;
    }
    v8f o[4];
#pragma unroll
    for (int nt = 0; nt < 4; ++nt) o[nt] = v8f_zero();
    bf16_t* myp = &pt[wv][0];
#pragma unroll
    for (int kt = 0; kt < 7; ++kt) {
#pragma unroll
      for (int tt = 0; tt < 2; ++tt) {
        int t = kt * 2 + tt;
#pragma unroll
        for (int r = 0; r < 8; ++r)
          myp[(r + half * 8) * 32 + tt * 16 + l16] = f2bf(s[t][r] * inv[r]);
      }
      Frag16 pf;
      {
        const bf16_t* p = myp + l16 * 32 + half * 8;
        pf.h[0] = *(const v8bf*)p;
        pf.h[1] = *(const v8bf*)(p + 16);
      }
#pragma unroll
      for (int nt = 0; nt < 4; ++nt) {
        Frag16 vf;
        const bf16_t* vp = vs + (nt * 16 + l16) * NPAD + kt * 32 + half * 8;
        vf.h[0] = *(const v8bf*)vp;
        vf.h[1] = *(const v8bf*)(vp + 16);
        o[nt] = __builtin_amdgcn_wmma_f32_16x16x32_bf16(
            false, pf.v, false, vf.v, (short)0, o[nt], false, false);
      }
    }
#pragma unroll
    for (int nt = 0; nt < 4; ++nt) {
      int dcol = nt * 16 + l16;
#pragma unroll
      for (int r = 0; r < 8; ++r) {
        int row = mt * 16 + r + half * 8;
        if (row < NTOK)
          ob[((size_t)(b * NTOK + row)) * E_DIM + h * HDIM + dcol] = f2bf(o[nt][r]);
      }
    }
  }
}

// ---------------------------------------------------------------------------
// Global average pool over tokens -> bf16 pooled [32, 768]
// ---------------------------------------------------------------------------
__global__ __launch_bounds__(256)
void pool_kernel(const bf16_t* __restrict__ ln, bf16_t* __restrict__ pooled)
{
  int b = blockIdx.x;
#pragma unroll
  for (int i = 0; i < 3; ++i) {
    int e = threadIdx.x + i * 256;
    float s = 0.f;
    for (int n = 0; n < NTOK; ++n)
      s += bf2f(ln[((size_t)(b * NTOK + n)) * E_DIM + e]);
    pooled[(size_t)b * E_DIM + e] = f2bf(s * (1.0f / NTOK));
  }
}

// ---------------------------------------------------------------------------
extern "C" void kernel_launch(void* const* d_in, const int* in_sizes, int n_in,
                              void* d_out, int out_size, void* d_ws, size_t ws_size,
                              hipStream_t stream)
{
  (void)in_sizes; (void)n_in; (void)out_size; (void)ws_size;
  const float* x      = (const float*)d_in[0];
  const float* conv_w = (const float*)d_in[1];
  const float* conv_b = (const float*)d_in[2];
  const float* h_emb  = (const float*)d_in[3];
  const float* w_emb  = (const float*)d_in[4];
  const float* ln1_w  = (const float*)d_in[5];
  const float* ln1_b  = (const float*)d_in[6];
  const float* qkv_w  = (const float*)d_in[7];
  const float* qn_w   = (const float*)d_in[8];
  const float* qn_b   = (const float*)d_in[9];
  const float* kn_w   = (const float*)d_in[10];
  const float* kn_b   = (const float*)d_in[11];
  const float* proj_w = (const float*)d_in[12];
  const float* proj_b = (const float*)d_in[13];
  const float* ln2_w  = (const float*)d_in[14];
  const float* ln2_b  = (const float*)d_in[15];
  const float* fc1_w  = (const float*)d_in[16];
  const float* fc1_b  = (const float*)d_in[17];
  const float* fc2_w  = (const float*)d_in[18];
  const float* fc2_b  = (const float*)d_in[19];
  const float* norm_w = (const float*)d_in[20];
  const float* norm_b = (const float*)d_in[21];
  const float* head_w = (const float*)d_in[22];
  const float* head_b = (const float*)d_in[23];

  char* base = (char*)d_ws;
  size_t off = 0;
  auto carve = [&](size_t bytes) -> void* {
    off = (off + 255) & ~(size_t)255;
    void* r = base + off;
    off += bytes;
    return r;
  };
  float*  tok    = (float*) carve((size_t)MROWS * E_DIM * 4);
  bf16_t* lnb    = (bf16_t*)carve((size_t)MROWS * E_DIM * 2);
  float*  qkvb   = (float*) carve((size_t)MROWS * 3 * E_DIM * 4);
  bf16_t* qn     = (bf16_t*)carve((size_t)BATCH * NHEAD * NPAD * HDIM * 2);
  bf16_t* kn     = (bf16_t*)carve((size_t)BATCH * NHEAD * NPAD * HDIM * 2);
  bf16_t* vnT    = (bf16_t*)carve((size_t)BATCH * NHEAD * HDIM * NPAD * 2);
  bf16_t* ob     = (bf16_t*)carve((size_t)MROWS * E_DIM * 2);
  bf16_t* ffb    = (bf16_t*)carve((size_t)MROWS * FFDIM * 2);
  bf16_t* pooled = (bf16_t*)carve((size_t)BATCH * E_DIM * 2);
  bf16_t* patches = ffb;  // alias: ffb unused before first fc1

  // bf16 weight mirrors (converted once per launch)
  const size_t qkvW = (size_t)3 * E_DIM * E_DIM;   // 1,769,472
  const size_t prjW = (size_t)E_DIM * E_DIM;       //   589,824
  const size_t fc1W = (size_t)FFDIM * E_DIM;       // 2,359,296
  const size_t fc2W = (size_t)E_DIM * FFDIM;       // 2,359,296
  bf16_t* conv_wb = (bf16_t*)carve((size_t)E_DIM * E_DIM * 2);
  bf16_t* qkv_wb  = (bf16_t*)carve(12 * qkvW * 2);
  bf16_t* proj_wb = (bf16_t*)carve(12 * prjW * 2);
  bf16_t* fc1_wb  = (bf16_t*)carve(12 * fc1W * 2);
  bf16_t* fc2_wb  = (bf16_t*)carve(12 * fc2W * 2);
  bf16_t* head_wb = (bf16_t*)carve((size_t)1000 * E_DIM * 2);

  auto cvt = [&](const float* s, bf16_t* d, size_t n) {
    long n4 = (long)(n / 4);
    cvt_f32_bf16<<<(unsigned)((n4 + 255) / 256), 256, 0, stream>>>(s, d, n4);
  };
  cvt(conv_w, conv_wb, (size_t)E_DIM * E_DIM);
  cvt(qkv_w,  qkv_wb,  12 * qkvW);
  cvt(proj_w, proj_wb, 12 * prjW);
  cvt(fc1_w,  fc1_wb,  12 * fc1W);
  cvt(fc2_w,  fc2_wb,  12 * fc2W);
  cvt(head_w, head_wb, (size_t)1000 * E_DIM);

  // ---- patch embed (gather + WMMA GEMM vs conv_w[768,768]) + pos embed
  patch_gather<<<MROWS, 256, 0, stream>>>(x, patches);
  {
    dim3 g(E_DIM / 256, MROWS / 128);
    gemm_wmma_bf16<0><<<g, 256, 0, stream>>>(patches, conv_wb, conv_b, nullptr,
                                             tok, nullptr, MROWS, E_DIM, E_DIM, 0);
  }
  pos_add<<<MROWS, 256, 0, stream>>>(tok, h_emb, w_emb);

  for (int d = 0; d < 12; ++d) {
    layernorm768<<<MROWS, 256, 0, stream>>>(tok, ln1_w + d * E_DIM, ln1_b + d * E_DIM, lnb);
    {
      dim3 g((3 * E_DIM) / 256, MROWS / 128);
      gemm_wmma_bf16<0><<<g, 256, 0, stream>>>(lnb, qkv_wb + (size_t)d * qkvW,
                                               nullptr, nullptr, qkvb, nullptr,
                                               MROWS, 3 * E_DIM, E_DIM, 0);
    }
    qkv_headnorm<<<BATCH * NPAD, 256, 0, stream>>>(qkvb,
        qn_w + d * HDIM, qn_b + d * HDIM, kn_w + d * HDIM, kn_b + d * HDIM,
        qn, kn, vnT);
    attention_wmma<<<dim3(NHEAD, BATCH), 128, 0, stream>>>(qn, kn, vnT, ob);
    {
      dim3 g(E_DIM / 256, MROWS / 128);
      gemm_wmma_bf16<0><<<g, 256, 0, stream>>>(ob, proj_wb + (size_t)d * prjW,
                                               proj_b + d * E_DIM, tok, tok, nullptr,
                                               MROWS, E_DIM, E_DIM, 0);
    }
    layernorm768<<<MROWS, 256, 0, stream>>>(tok, ln2_w + d * E_DIM, ln2_b + d * E_DIM, lnb);
    {
      dim3 g(FFDIM / 256, MROWS / 128);
      gemm_wmma_bf16<0><<<g, 256, 0, stream>>>(lnb, fc1_wb + (size_t)d * fc1W,
                                               fc1_b + d * FFDIM, nullptr, nullptr, ffb,
                                               MROWS, FFDIM, E_DIM, 1);
    }
    {
      dim3 g(E_DIM / 256, MROWS / 128);
      gemm_wmma_bf16<0><<<g, 256, 0, stream>>>(ffb, fc2_wb + (size_t)d * fc2W,
                                               fc2_b + d * E_DIM, tok, tok, nullptr,
                                               MROWS, E_DIM, FFDIM, 0);
    }
  }

  layernorm768<<<MROWS, 256, 0, stream>>>(tok, norm_w, norm_b, lnb);
  pool_kernel<<<BATCH, 256, 0, stream>>>(lnb, pooled);
  {
    dim3 g((1000 + 255) / 256, 1);
    gemm_wmma_bf16<1><<<g, 256, 0, stream>>>(pooled, head_wb, head_b, nullptr,
                                             (float*)d_out, nullptr, BATCH, 1000, E_DIM, 0);
  }
}
